// MetaMultiHeadedAttention_4561255268488
// MI455X (gfx1250) — matrix operand order repaired and checked by reading, not compiled, not run on hardware
//
#include <hip/hip_runtime.h>
#include <hip/hip_bf16.h>

// MetaMultiHeadedAttention for MI455X (gfx1250, wave32, WMMA).
// BS=4, L=1024, DM=512, H=8, P=512, DLANG=64, DK=64.

#define BSZ    4
#define LSEQ   1024
#define DMODEL 512
#define NH     8
#define NP     512
#define DLANG  64
#define DKH    64

typedef __bf16 bf16_t;
typedef bf16_t v16bf __attribute__((ext_vector_type(16)));
typedef float  v8f   __attribute__((ext_vector_type(8)));

// ---------- WMMA fragment loaders (contiguous k order, same for A and B) ----------
// A matrix 16x32: lane holds row m=lane&15; elem e -> k=e+16*(lane>>4)
// B matrix 32x16: lane holds col n=lane&15; elem e -> k=e+16*(lane>>4)

__device__ __forceinline__ v16bf load_a16x32_f32(const float* __restrict__ base,
                                                 int ld, int lane) {
  int m = lane & 15, hlf = lane >> 4;
  v16bf a;
#pragma unroll
  for (int e = 0; e < 16; ++e) {
    int k = e + (hlf << 4);
    a[e] = (bf16_t)base[(size_t)m * ld + k];
  }
  return a;
}

__device__ __forceinline__ v16bf load_a16x32_bf(const bf16_t* __restrict__ base,
                                                int ld, int lane) {
  int m = lane & 15, hlf = lane >> 4;
  v16bf a;
#pragma unroll
  for (int e = 0; e < 16; ++e) {
    int k = e + (hlf << 4);
    a[e] = base[(size_t)m * ld + k];
  }
  return a;
}

__device__ __forceinline__ v16bf load_b32x16_rm_f32(const float* __restrict__ base,
                                                    int ld, int lane) {
  int n = lane & 15, hlf = lane >> 4;
  v16bf b;
#pragma unroll
  for (int e = 0; e < 16; ++e) {
    int k = e + (hlf << 4);
    b[e] = (bf16_t)base[(size_t)k * ld + n];
  }
  return b;
}

__device__ __forceinline__ v16bf load_b32x16_rm_bf(const bf16_t* __restrict__ base,
                                                   int ld, int lane) {
  int n = lane & 15, hlf = lane >> 4;
  v16bf b;
#pragma unroll
  for (int e = 0; e < 16; ++e) {
    int k = e + (hlf << 4);
    b[e] = base[(size_t)k * ld + n];
  }
  return b;
}

// B = M^T where M is row-major N x K: B[k][n] = base[n*ld + k]
__device__ __forceinline__ v16bf load_b32x16_tr_bf(const bf16_t* __restrict__ base,
                                                   int ld, int lane) {
  int n = lane & 15, hlf = lane >> 4;
  v16bf b;
#pragma unroll
  for (int e = 0; e < 16; ++e) {
    int k = e + (hlf << 4);
    b[e] = base[(size_t)n * ld + k];
  }
  return b;
}

__device__ __forceinline__ v16bf load_b32x16_tr_f32(const float* __restrict__ base,
                                                    int ld, int lane) {
  int n = lane & 15, hlf = lane >> 4;
  v16bf b;
#pragma unroll
  for (int e = 0; e < 16; ++e) {
    int k = e + (hlf << 4);
    b[e] = (bf16_t)base[(size_t)n * ld + k];
  }
  return b;
}

__device__ __forceinline__ v8f wmma_bf16(v16bf a, v16bf b, v8f c) {
  return __builtin_amdgcn_wmma_f32_16x16x32_bf16(false, a, false, b,
                                                 (short)0, c, false, false);
}

// ---------- Kernel 1: hypernetwork weight generation (streaming, HBM-bound) ----------
// W[b][i][o] = sum_d lang[b][d] * Wg[d][i][o]; each Wg element read exactly once.
__global__ __launch_bounds__(256) void gen_weights_kernel(
    const float* __restrict__ Wg, const float* __restrict__ lang,
    float* __restrict__ Wout) {
  __shared__ float sl[BSZ * DLANG];  // 256 floats
  int tid = threadIdx.x;
  sl[tid] = lang[tid];
  __syncthreads();
  size_t io = (size_t)blockIdx.x * 256 + tid;  // < DMODEL*DMODEL
  float acc[BSZ];
#pragma unroll
  for (int b = 0; b < BSZ; ++b) acc[b] = 0.f;
  for (int d = 0; d < DLANG; ++d) {
    if (d + 8 < DLANG)  // hide the 1MB-strided d-walk latency (global_prefetch_b8)
      __builtin_prefetch(Wg + (size_t)(d + 8) * (DMODEL * DMODEL) + io, 0, 3);
    float w = Wg[(size_t)d * (DMODEL * DMODEL) + io];
#pragma unroll
    for (int b = 0; b < BSZ; ++b) acc[b] += sl[b * DLANG + d] * w;
  }
#pragma unroll
  for (int b = 0; b < BSZ; ++b)
    Wout[(size_t)b * (DMODEL * DMODEL) + io] = acc[b];
}

__global__ __launch_bounds__(256) void gen_bias_kernel(
    const float* __restrict__ Bg, const float* __restrict__ lang,
    float* __restrict__ bout) {
  int o = blockIdx.x * blockDim.x + threadIdx.x;
  if (o >= DMODEL) return;
#pragma unroll
  for (int b = 0; b < BSZ; ++b) {
    float acc = 0.f;
    for (int d = 0; d < DLANG; ++d) acc += lang[b * DLANG + d] * Bg[d * DMODEL + o];
    bout[b * DMODEL + o] = acc;
  }
}

// ---------- Kernel 2: meta-linear apply, WMMA bf16 GEMM ----------
// Y[b] = X[b] (L x DM) @ W[b] (DM x DM) + bias[b].
// One wave computes a 16x64 tile: A fragment reused across 4 WMMAs per k-step.
// OutT = bf16 for the q/k/v projections (consumed only as WMMA operands),
// OutT = float for the final output projection.
template <typename OutT>
__global__ __launch_bounds__(32) void meta_apply_kernel(
    const float* __restrict__ X, const float* __restrict__ W,
    const float* __restrict__ bias, OutT* __restrict__ Y) {
  const int Mdim = LSEQ, Kdim = DMODEL, Ndim = DMODEL;
  int lane = threadIdx.x;
  int n0 = blockIdx.x * 64, m0 = blockIdx.y * 16, b = blockIdx.z;
  const float* Xb = X + ((size_t)b * Mdim + m0) * Kdim;
  const float* Wb = W + (size_t)b * Kdim * Ndim + n0;
  v8f c[4] = {};
  for (int k0 = 0; k0 < Kdim; k0 += 32) {
    v16bf a = load_a16x32_f32(Xb + k0, Kdim, lane);
    const float* wrow = Wb + (size_t)k0 * Ndim;
#pragma unroll
    for (int t = 0; t < 4; ++t) {
      v16bf bb = load_b32x16_rm_f32(wrow + t * 16, Ndim, lane);
      c[t] = wmma_bf16(a, bb, c[t]);
    }
  }
  int hlf = lane >> 4, n = lane & 15;
#pragma unroll
  for (int t = 0; t < 4; ++t) {
    float bvv = bias[b * Ndim + n0 + t * 16 + n];
#pragma unroll
    for (int r = 0; r < 8; ++r) {
      int m = m0 + r + hlf * 8;
      Y[((size_t)b * Mdim + m) * Ndim + n0 + t * 16 + n] = (OutT)(c[t][r] + bvv);
    }
  }
}

// ---------- Kernel 3: fused relative-position flash attention ----------
// One wave handles 16 query rows of one (b,h). Q/K/V arrive pre-converted to
// bf16. LDS holds qr (16xP f32) and the attn_p bucket accumulator (16xP f32);
// online softmax; all matmuls via WMMA.
__global__ __launch_bounds__(32) void attn_kernel(
    const bf16_t* __restrict__ Q, const bf16_t* __restrict__ Kc,
    const bf16_t* __restrict__ V, const float* __restrict__ r_k,
    const float* __restrict__ r_v, const int* __restrict__ path_map,
    const float* __restrict__ ap, const unsigned char* __restrict__ mask,
    float* __restrict__ ctx) {
  __shared__ float s_qr[16 * NP];    // q . r_k^T
  __shared__ float s_acc[16 * NP];   // attn_p bucket accumulator (unnormalized)
  __shared__ float s_pt[16 * 32];    // prob tile staging (C-layout -> A-layout)
  __shared__ float s_alpha[16];      // per-row online-softmax rescale

  int lane = threadIdx.x;
  int hlf = lane >> 4, n = lane & 15;
  int i0 = blockIdx.x * 16, h = blockIdx.y, b = blockIdx.z;

  const bf16_t* qb = Q + ((size_t)b * LSEQ + i0) * DMODEL + h * DKH;
  v16bf aq0 = load_a16x32_bf(qb, DMODEL, lane);        // dk 0..31
  v16bf aq1 = load_a16x32_bf(qb + 32, DMODEL, lane);   // dk 32..63

  // qr tile: (16 x 64) @ (64 x P) -> LDS
  const float* rkb = r_k + (size_t)(b * NH + h) * NP * DKH;
  for (int p0 = 0; p0 < NP; p0 += 16) {
    v16bf b0 = load_b32x16_tr_f32(rkb + (size_t)p0 * DKH, DKH, lane);
    v16bf b1 = load_b32x16_tr_f32(rkb + (size_t)p0 * DKH + 32, DKH, lane);
    v8f c = {};
    c = wmma_bf16(aq0, b0, c);
    c = wmma_bf16(aq1, b1, c);
#pragma unroll
    for (int r = 0; r < 8; ++r) s_qr[(r + hlf * 8) * NP + p0 + n] = c[r];
  }
  for (int idx = lane; idx < 16 * NP; idx += 32) s_acc[idx] = 0.f;
  __syncthreads();

  float m_run[8], l_run[8];
#pragma unroll
  for (int r = 0; r < 8; ++r) { m_run[r] = -3.0e38f; l_run[r] = 0.f; }
  v8f xc[4] = {};  // x accumulator, 16 x 64 (4 N-tiles)

  const float scale = 0.125f;  // 1/sqrt(64)
  for (int j0 = 0; j0 < LSEQ; j0 += 32) {
    // ---- scores: two 16x16 tiles (columns j0..j0+31) ----
    v8f s0, s1;
    {
      const bf16_t* kb0 = Kc + ((size_t)b * LSEQ + j0) * DMODEL + h * DKH;
      v8f c = {};
      c = wmma_bf16(aq0, load_b32x16_tr_bf(kb0, DMODEL, lane), c);
      c = wmma_bf16(aq1, load_b32x16_tr_bf(kb0 + 32, DMODEL, lane), c);
      s0 = c;
      const bf16_t* kb1 = Kc + ((size_t)b * LSEQ + j0 + 16) * DMODEL + h * DKH;
      v8f c2 = {};
      c2 = wmma_bf16(aq0, load_b32x16_tr_bf(kb1, DMODEL, lane), c2);
      c2 = wmma_bf16(aq1, load_b32x16_tr_bf(kb1 + 32, DMODEL, lane), c2);
      s1 = c2;
    }
    // ---- add gathered qr, scale, +ap, mask ----
    int idx0[8], idx1[8];
#pragma unroll
    for (int r = 0; r < 8; ++r) {
      int row = r + hlf * 8;
      int i = i0 + row;
      size_t rbase = ((size_t)b * LSEQ + i) * LSEQ;
      size_t abase = ((size_t)(b * NH + h) * LSEQ + i) * LSEQ;
      int ja = j0 + n, jb = j0 + 16 + n;
      idx0[r] = path_map[rbase + ja];
      idx1[r] = path_map[rbase + jb];
      float va = (s0[r] + s_qr[row * NP + idx0[r]]) * scale + ap[abase + ja];
      float vb2 = (s1[r] + s_qr[row * NP + idx1[r]]) * scale + ap[abase + jb];
      if (!mask[rbase + ja]) va = -1e9f;
      if (!mask[rbase + jb]) vb2 = -1e9f;
      s0[r] = va; s1[r] = vb2;
    }
    // ---- online softmax stats (reduce across 16-lane half-groups) ----
    float alpha[8], p0v[8], p1v[8];
#pragma unroll
    for (int r = 0; r < 8; ++r) {
      float mx = fmaxf(s0[r], s1[r]);
      for (int off = 1; off < 16; off <<= 1) mx = fmaxf(mx, __shfl_xor(mx, off, 32));
      float m_new = fmaxf(m_run[r], mx);
      float a = __expf(m_run[r] - m_new);
      float pa = __expf(s0[r] - m_new);
      float pb = __expf(s1[r] - m_new);
      float rs = pa + pb;
      for (int off = 1; off < 16; off <<= 1) rs += __shfl_xor(rs, off, 32);
      l_run[r] = l_run[r] * a + rs;
      m_run[r] = m_new;
      alpha[r] = a; p0v[r] = pa; p1v[r] = pb;
    }
    // ---- rescale accumulators (register x-frags + LDS attn_p) ----
#pragma unroll
    for (int t = 0; t < 4; ++t)
#pragma unroll
      for (int r = 0; r < 8; ++r) xc[t][r] *= alpha[r];
    if (n == 0) {
#pragma unroll
      for (int r = 0; r < 8; ++r) s_alpha[r + hlf * 8] = alpha[r];
    }
    __syncthreads();
    for (int idx = lane; idx < 16 * NP; idx += 32) s_acc[idx] *= s_alpha[idx >> 9];
    __syncthreads();
    // ---- scatter probs into buckets (ds_add_f32) + stage for A-frag ----
#pragma unroll
    for (int r = 0; r < 8; ++r) {
      int row = r + hlf * 8;
      atomicAdd(&s_acc[row * NP + idx0[r]], p0v[r]);
      atomicAdd(&s_acc[row * NP + idx1[r]], p1v[r]);
      s_pt[row * 32 + n] = p0v[r];
      s_pt[row * 32 + 16 + n] = p1v[r];
    }
    __syncthreads();
    // ---- x += p (16x32) @ v (32x64) ----
    v16bf apf;
#pragma unroll
    for (int e = 0; e < 16; ++e) {
      int k = e + (hlf << 4);
      apf[e] = (bf16_t)s_pt[n * 32 + k];  // row = lane&15
    }
    const bf16_t* vb = V + ((size_t)b * LSEQ + j0) * DMODEL + h * DKH;
#pragma unroll
    for (int t = 0; t < 4; ++t) {
      v16bf bv = load_b32x16_rm_bf(vb + t * 16, DMODEL, lane);
      xc[t] = wmma_bf16(apf, bv, xc[t]);
    }
    __syncthreads();
  }

  // ---- x += attn_p (16xP) @ r_v (P x 64), still unnormalized ----
  __syncthreads();
  const float* rvb = r_v + (size_t)(b * NH + h) * NP * DKH;
  for (int p0 = 0; p0 < NP; p0 += 32) {
    v16bf apf;
#pragma unroll
    for (int e = 0; e < 16; ++e) {
      int k = e + (hlf << 4);
      apf[e] = (bf16_t)s_acc[n * NP + p0 + k];
    }
#pragma unroll
    for (int t = 0; t < 4; ++t) {
      v16bf bv = load_b32x16_rm_f32(rvb + (size_t)p0 * DKH + t * 16, DKH, lane);
      xc[t] = wmma_bf16(apf, bv, xc[t]);
    }
  }
  // ---- normalize by 1/l and write context (BS, L, DM) f32 ----
#pragma unroll
  for (int r = 0; r < 8; ++r) {
    float inv_l = 1.0f / l_run[r];
    int i = i0 + r + hlf * 8;
    float* o = ctx + ((size_t)b * LSEQ + i) * DMODEL + h * DKH;
#pragma unroll
    for (int t = 0; t < 4; ++t) o[t * 16 + n] = xc[t][r] * inv_l;
  }
}

// ---------- launch ----------
extern "C" void kernel_launch(void* const* d_in, const int* in_sizes, int n_in,
                              void* d_out, int out_size, void* d_ws, size_t ws_size,
                              hipStream_t stream) {
  (void)in_sizes; (void)n_in; (void)out_size; (void)ws_size;
  const float* query = (const float*)d_in[0];
  const float* key   = (const float*)d_in[1];
  const float* value = (const float*)d_in[2];
  const float* lang  = (const float*)d_in[3];
  const unsigned char* mask = (const unsigned char*)d_in[4];
  const float* r_k   = (const float*)d_in[5];
  const float* r_v   = (const float*)d_in[6];
  const int* path_map = (const int*)d_in[7];
  const float* ap    = (const float*)d_in[8];
  const float* Wg_q  = (const float*)d_in[9];
  const float* Bg_q  = (const float*)d_in[10];
  const float* Wg_k  = (const float*)d_in[11];
  const float* Bg_k  = (const float*)d_in[12];
  const float* Wg_v  = (const float*)d_in[13];
  const float* Bg_v  = (const float*)d_in[14];
  const float* Wg_o  = (const float*)d_in[15];
  const float* Bg_o  = (const float*)d_in[16];

  const size_t Wsz = (size_t)BSZ * DMODEL * DMODEL;  // per-projection weights (f32)
  const size_t bsz = (size_t)BSZ * DMODEL;
  const size_t Xsz = (size_t)BSZ * LSEQ * DMODEL;

  char* wsb = (char*)d_ws;
  float* Wq = (float*)wsb;  wsb += Wsz * sizeof(float);
  float* Wk = (float*)wsb;  wsb += Wsz * sizeof(float);
  float* Wv = (float*)wsb;  wsb += Wsz * sizeof(float);
  float* Wo = (float*)wsb;  wsb += Wsz * sizeof(float);
  float* bq = (float*)wsb;  wsb += bsz * sizeof(float);
  float* bk = (float*)wsb;  wsb += bsz * sizeof(float);
  float* bv = (float*)wsb;  wsb += bsz * sizeof(float);
  float* bo = (float*)wsb;  wsb += bsz * sizeof(float);
  bf16_t* Qb = (bf16_t*)wsb; wsb += Xsz * sizeof(bf16_t);
  bf16_t* Kb = (bf16_t*)wsb; wsb += Xsz * sizeof(bf16_t);
  bf16_t* Vb = (bf16_t*)wsb; wsb += Xsz * sizeof(bf16_t);
  float* Cb = (float*)wsb;

  dim3 gw(DMODEL * DMODEL / 256);
  gen_weights_kernel<<<gw, 256, 0, stream>>>(Wg_q, lang, Wq);
  gen_weights_kernel<<<gw, 256, 0, stream>>>(Wg_k, lang, Wk);
  gen_weights_kernel<<<gw, 256, 0, stream>>>(Wg_v, lang, Wv);
  gen_weights_kernel<<<gw, 256, 0, stream>>>(Wg_o, lang, Wo);
  gen_bias_kernel<<<2, 256, 0, stream>>>(Bg_q, lang, bq);
  gen_bias_kernel<<<2, 256, 0, stream>>>(Bg_k, lang, bk);
  gen_bias_kernel<<<2, 256, 0, stream>>>(Bg_v, lang, bv);
  gen_bias_kernel<<<2, 256, 0, stream>>>(Bg_o, lang, bo);

  dim3 gg(DMODEL / 64, LSEQ / 16, BSZ);
  meta_apply_kernel<bf16_t><<<gg, 32, 0, stream>>>(query, Wq, bq, Qb);
  meta_apply_kernel<bf16_t><<<gg, 32, 0, stream>>>(key,   Wk, bk, Kb);
  meta_apply_kernel<bf16_t><<<gg, 32, 0, stream>>>(value, Wv, bv, Vb);

  attn_kernel<<<dim3(LSEQ / 16, NH, BSZ), 32, 0, stream>>>(
      Qb, Kb, Vb, r_k, r_v, path_map, ap, mask, Cb);

  meta_apply_kernel<float><<<gg, 32, 0, stream>>>(Cb, Wo, bo, (float*)d_out);
}
